// Res_EGblk_0_687194767629
// MI455X (gfx1250) — matrix operand order repaired and checked by reading, not compiled, not run on hardware
//
#include <hip/hip_runtime.h>
#include <hip/hip_bf16.h>
#include <stdint.h>

typedef __attribute__((ext_vector_type(16))) _Float16 v16h;
typedef __attribute__((ext_vector_type(8)))  float    v8f;

#define N_NODES 50000
#define N_EDGES 800000
#define DN      128      // D_NODE
#define DE      64       // D_EDGE
#define DF      16       // D_EF
#define KIN     208      // IN_CH
#define KPAD    224      // padded K for lin2 WMMA
#define FS      228      // LDS feat row stride (halves), padded vs bank conflicts
#define EPSV    1e-5f

// ---------------------------------------------------------------------------
// elementwise f32 -> f16 convert
__global__ __launch_bounds__(256) void k_cvt_f16(const float* __restrict__ s,
                                                 _Float16* __restrict__ d, int n) {
    int i = blockIdx.x * 256 + threadIdx.x;
    if (i < n) d[i] = (_Float16)s[i];
}

// ---------------------------------------------------------------------------
// pack a row-major [K x Nn] f32 weight matrix into WMMA B-fragment layout.
// B 32x16 (KxN) f16 frag in this code's contiguous k order (A uses the same): lane = half*16 + n ; VGPR r holds K = half*16+2r,+1.
// packed[((ks*NT + nt)*32 + lane)*16 + 2r + j] so a wave loads one v16h / frag.
__global__ __launch_bounds__(64) void k_pack_b(const float* __restrict__ W,
                                               _Float16* __restrict__ out,
                                               int K, int Nn, int K32, int NT) {
    int gid = blockIdx.x * 64 + threadIdx.x;
    int total = K32 * NT * 32;
    if (gid >= total) return;
    int lane = gid & 31;
    int nt   = (gid >> 5) % NT;
    int ks   = gid / (32 * NT);
    int nl   = lane & 15, half = lane >> 4;
    int col  = nt * 16 + nl;
    _Float16* o = out + (size_t)gid * 16;
#pragma unroll
    for (int r = 0; r < 8; ++r) {
        int k = ks * 32 + half * 16 + 2 * r;
#pragma unroll
        for (int j = 0; j < 2; ++j) {
            int kk = k + j;
            float v = (kk < K && col < Nn) ? W[(size_t)kk * Nn + col] : 0.0f;
            o[2 * r + j] = (_Float16)v;
        }
    }
}

// ---------------------------------------------------------------------------
// zero / init scratch that is accumulated into (graph-replay safe: runs every call)
__global__ __launch_bounds__(256) void k_init(float* __restrict__ agg,
                                              float* __restrict__ deg,
                                              float* __restrict__ ebn,
                                              float* __restrict__ nbn) {
    int i = blockIdx.x * 256 + threadIdx.x;
    if (i < N_NODES * DN) agg[i] = 0.0f;
    if (i < N_NODES)      deg[i] = 1.0f;     // self-loop weight
    if (i < 2 * DE)       ebn[i] = 0.0f;     // edge BN sum / sumsq
    if (i < 2 * DN)       nbn[i] = 0.0f;     // node BN sum / sumsq
}

// ---------------------------------------------------------------------------
// dual node GEMM: h = x@lin1_w + lin1_b (stored f16), xw = x@gcn_w (f32)
// grid.x = N/16 tiles, 8 waves per block, wave = output ntile (shared A frag)
__global__ __launch_bounds__(256) void k_node_gemm(const _Float16* __restrict__ x16,
                                                   const _Float16* __restrict__ pb1,
                                                   const _Float16* __restrict__ pbg,
                                                   const float* __restrict__ lin1_b,
                                                   _Float16* __restrict__ h16,
                                                   float* __restrict__ xw) {
    int tile = blockIdx.x;
    int wv   = threadIdx.x >> 5;      // ntile 0..7
    int lane = threadIdx.x & 31;
    int nl   = lane & 15, half = lane >> 4;
    int row0 = tile * 16;
    const _Float16* arow = x16 + (size_t)(row0 + nl) * DN;  // A row m = nl
    v8f c1 = {}; v8f c2 = {};
#pragma unroll
    for (int ks = 0; ks < 4; ++ks) {
        v16h a;
#pragma unroll
        for (int r = 0; r < 8; ++r) {
            int k0 = ks * 32 + half * 16 + r * 2;
            a[2 * r]     = arow[k0];
            a[2 * r + 1] = arow[k0 + 1];
        }
        v16h b1 = *(const v16h*)(pb1 + (size_t)(((ks * 8 + wv) * 32 + lane)) * 16);
        v16h b2 = *(const v16h*)(pbg + (size_t)(((ks * 8 + wv) * 32 + lane)) * 16);
        c1 = __builtin_amdgcn_wmma_f32_16x16x32_f16(false, a, false, b1, (short)0, c1, false, false);
        c2 = __builtin_amdgcn_wmma_f32_16x16x32_f16(false, a, false, b2, (short)0, c2, false, false);
    }
    int col = wv * 16 + nl;
    float bias = lin1_b[col];
#pragma unroll
    for (int r = 0; r < 8; ++r) {
        int row = row0 + half * 8 + r;
        h16[(size_t)row * DN + col] = (_Float16)(c1[r] + bias);
        xw [(size_t)row * DN + col] = c2[r];
    }
}

// ---------------------------------------------------------------------------
// edge GEMM: feat = [relu(h[src]+h[dst]) | edge_attr | edge_f | 0pad] (16 x 224)
// ef_pre = feat @ lin2_w + lin2_b   -> written into d_out ef region (scratch)
// one block (4 waves) per 16-edge tile; LDS staging shared by the 4 ntiles.
// h-row gather uses gfx1250 GLOBAL_LOAD_ASYNC_TO_LDS (ASYNCcnt) so it flies
// while the edge_attr/edge_f f32->f16 staging executes; each thread touches
// only LDS it requested itself, so a wave-local s_wait_asynccnt 0 is enough
// before the block barrier.
__global__ __launch_bounds__(128) void k_edge_gemm(const _Float16* __restrict__ h16,
                                                   const int* __restrict__ eidx,
                                                   const float* __restrict__ edge_attr,
                                                   const float* __restrict__ edge_f,
                                                   const _Float16* __restrict__ pb2,
                                                   const float* __restrict__ lin2_b,
                                                   float* __restrict__ ef_pre) {
    __shared__ _Float16 hs[16 * DN];       // gathered h[src] rows
    __shared__ _Float16 hd[16 * DN];       // gathered h[dst] rows
    __shared__ _Float16 featA[16 * FS];
    int e0 = blockIdx.x * 16;
    int t  = threadIdx.x;
    int e  = t & 15, seg = t >> 4;          // seg 0..7
    int ge = e0 + e;
    int s  = eidx[ge];
    int d  = eidx[N_EDGES + ge];

    // async gather: 32B per thread from each of the two h rows -> LDS staging
    const _Float16* gs = h16 + (size_t)s * DN + seg * 16;
    const _Float16* gd = h16 + (size_t)d * DN + seg * 16;
    uint32_t ls = (uint32_t)(uintptr_t)(hs + e * DN + seg * 16);
    uint32_t ld = (uint32_t)(uintptr_t)(hd + e * DN + seg * 16);
    asm volatile(
        "global_load_async_to_lds_b128 %0, %2, off\n\t"
        "global_load_async_to_lds_b128 %0, %2, off offset:16\n\t"
        "global_load_async_to_lds_b128 %1, %3, off\n\t"
        "global_load_async_to_lds_b128 %1, %3, off offset:16"
        :: "v"(ls), "v"(ld), "v"(gs), "v"(gd)
        : "memory");

    // overlap: stage edge_attr / edge_f / zero-pad (independent of the gather)
    _Float16* fr = featA + e * FS;
    const float* ar = edge_attr + (size_t)ge * DE + seg * 8;
#pragma unroll
    for (int i = 0; i < 8; ++i) fr[DN + seg * 8 + i] = (_Float16)ar[i];
    if (seg < 2) {
        const float* er = edge_f + (size_t)ge * DF + seg * 8;
#pragma unroll
        for (int i = 0; i < 8; ++i) fr[DN + DE + seg * 8 + i] = (_Float16)er[i];
    } else if (seg < 4) {
#pragma unroll
        for (int i = 0; i < 8; ++i) fr[KIN + (seg - 2) * 8 + i] = (_Float16)0;
    }

    // wait for this wave's async transfers, then x_em = relu(h_src + h_dst)
    asm volatile("s_wait_asynccnt 0x0" ::: "memory");
    const _Float16* ps = hs + e * DN + seg * 16;
    const _Float16* pd = hd + e * DN + seg * 16;
#pragma unroll
    for (int i = 0; i < 16; ++i) {
        _Float16 v = ps[i] + pd[i];
        fr[seg * 16 + i] = v > (_Float16)0 ? v : (_Float16)0;
    }
    __syncthreads();

    int wv = t >> 5, lane = t & 31;
    int nl = lane & 15, half = lane >> 4;
    const _Float16* am = featA + nl * FS;
    v8f c = {};
#pragma unroll
    for (int ks = 0; ks < 7; ++ks) {
        v16h a;
#pragma unroll
        for (int r = 0; r < 8; ++r) {
            int k0 = ks * 32 + half * 16 + r * 2;
            a[2 * r]     = am[k0];
            a[2 * r + 1] = am[k0 + 1];
        }
        v16h b = *(const v16h*)(pb2 + (size_t)(((ks * 4 + wv) * 32 + lane)) * 16);
        c = __builtin_amdgcn_wmma_f32_16x16x32_f16(false, a, false, b, (short)0, c, false, false);
    }
    int col = wv * 16 + nl;
    float bias = lin2_b[col];
#pragma unroll
    for (int r = 0; r < 8; ++r) {
        int m = half * 8 + r;
        ef_pre[(size_t)(e0 + m) * DE + col] = c[r] + bias;
    }
}

// ---------------------------------------------------------------------------
// edge batchnorm stats: per-column sum / sumsq over E rows
__global__ __launch_bounds__(256) void k_ebn_stats(const float* __restrict__ efp,
                                                   float* __restrict__ sums) {
    __shared__ float s1[DE], s2[DE];
    int t = threadIdx.x;
    if (t < DE) { s1[t] = 0.f; s2[t] = 0.f; }
    __syncthreads();
    int c  = t & 63;
    int r0 = blockIdx.x * 4 + (t >> 6);
    float a = 0.f, b = 0.f;
    for (int r = r0; r < N_EDGES; r += gridDim.x * 4) {
        float v = efp[(size_t)r * DE + c];
        a += v; b += v * v;
    }
    atomicAdd(&s1[c], a);
    atomicAdd(&s2[c], b);
    __syncthreads();
    if (t < DE) {
        atomicAdd(&sums[t], s1[t]);
        atomicAdd(&sums[DE + t], s2[t]);
    }
}

__global__ void k_ebn_final(const float* __restrict__ sums, const float* __restrict__ g,
                            const float* __restrict__ b, float* __restrict__ scsh) {
    int c = threadIdx.x;
    if (c >= DE) return;
    float mu  = sums[c] / (float)N_EDGES;
    float var = sums[DE + c] / (float)N_EDGES - mu * mu;
    float rs  = rsqrtf(var + EPSV) * g[c];
    scsh[c]      = rs;
    scsh[DE + c] = b[c] - mu * rs;
}

// ---------------------------------------------------------------------------
// edge pass 2: BN + relu, edge weight w = relu(ef @ linw_w + b), ef_out = ef + edge_attr
// (reads and rewrites d_out ef region in place)
__global__ __launch_bounds__(256) void k_edge_pass2(float* __restrict__ efreg,
                                                    const float* __restrict__ edge_attr,
                                                    const float* __restrict__ scsh,
                                                    const float* __restrict__ lww,
                                                    const float* __restrict__ lwb,
                                                    float* __restrict__ w) {
    int e = blockIdx.x * 256 + threadIdx.x;
    if (e >= N_EDGES) return;
    float* row      = efreg + (size_t)e * DE;
    const float* ar = edge_attr + (size_t)e * DE;
    __builtin_prefetch(row + 64 * DE, 0, 1);
    float acc = 0.f;
#pragma unroll
    for (int c = 0; c < DE; ++c) {
        float v = fmaxf(row[c] * scsh[c] + scsh[DE + c], 0.f);
        acc += v * lww[c];
        row[c] = v + ar[c];
    }
    w[e] = fmaxf(acc + lwb[0], 0.f);
}

// ---------------------------------------------------------------------------
__global__ __launch_bounds__(256) void k_degree(const int* __restrict__ eidx,
                                                const float* __restrict__ w,
                                                float* __restrict__ deg) {
    int e = blockIdx.x * 256 + threadIdx.x;
    if (e >= N_EDGES) return;
    atomicAdd(&deg[eidx[N_EDGES + e]], w[e]);
}

__global__ __launch_bounds__(256) void k_dinv(const float* __restrict__ deg,
                                              float* __restrict__ dinv) {
    int i = blockIdx.x * 256 + threadIdx.x;
    if (i >= N_NODES) return;
    float d = deg[i];
    dinv[i] = d > 0.f ? rsqrtf(d) : 0.f;
}

// scatter-add: agg[dst] += dinv[src]*w*dinv[dst] * xw[src]   (32 lanes x 4 floats / edge)
__global__ __launch_bounds__(256) void k_scatter(const int* __restrict__ eidx,
                                                 const float* __restrict__ w,
                                                 const float* __restrict__ dinv,
                                                 const float* __restrict__ xw,
                                                 float* __restrict__ agg) {
    int gid = blockIdx.x * 256 + threadIdx.x;
    int e = gid >> 5;
    if (e >= N_EDGES) return;
    int lane = gid & 31;
    int s = eidx[e], d = eidx[N_EDGES + e];
    float nm = dinv[s] * w[e] * dinv[d];
    float4 xv = *(const float4*)(xw + (size_t)s * DN + lane * 4);
    float* ap = agg + (size_t)d * DN + lane * 4;
    atomicAdd(ap + 0, nm * xv.x);
    atomicAdd(ap + 1, nm * xv.y);
    atomicAdd(ap + 2, nm * xv.z);
    atomicAdd(ap + 3, nm * xv.w);
}

// ---------------------------------------------------------------------------
// x1_pre = agg + dinv^2 * xw + gcn_b  (in place in agg) + node BN stats
__global__ __launch_bounds__(256) void k_node_pre(float* __restrict__ agg,
                                                  const float* __restrict__ xw,
                                                  const float* __restrict__ dinv,
                                                  const float* __restrict__ gcn_b,
                                                  float* __restrict__ nsums) {
    __shared__ float s1[DN], s2[DN];
    int t = threadIdx.x;
    if (t < DN) { s1[t] = 0.f; s2[t] = 0.f; }
    __syncthreads();
    int c  = t & 127;
    int r0 = blockIdx.x * 2 + (t >> 7);
    float a = 0.f, b = 0.f;
    float gb = gcn_b[c];
    for (int r = r0; r < N_NODES; r += gridDim.x * 2) {
        float di = dinv[r];
        size_t i = (size_t)r * DN + c;
        float v = agg[i] + di * di * xw[i] + gb;
        agg[i] = v;
        a += v; b += v * v;
    }
    atomicAdd(&s1[c], a);
    atomicAdd(&s2[c], b);
    __syncthreads();
    if (t < DN) {
        atomicAdd(&nsums[t], s1[t]);
        atomicAdd(&nsums[DN + t], s2[t]);
    }
}

__global__ void k_nbn_final(const float* __restrict__ sums, const float* __restrict__ g,
                            const float* __restrict__ b, float* __restrict__ scsh) {
    int c = threadIdx.x;
    if (c >= DN) return;
    float mu  = sums[c] / (float)N_NODES;
    float var = sums[DN + c] / (float)N_NODES - mu * mu;
    float rs  = rsqrtf(var + EPSV) * g[c];
    scsh[c]      = rs;
    scsh[DN + c] = b[c] - mu * rs;
}

__global__ __launch_bounds__(256) void k_node_out(const float* __restrict__ x1pre,
                                                  const float* __restrict__ x,
                                                  const float* __restrict__ scsh,
                                                  float* __restrict__ out) {
    int i = blockIdx.x * 256 + threadIdx.x;
    if (i >= N_NODES * DN) return;
    int c = i & 127;
    float v = fmaxf(x1pre[i] * scsh[c] + scsh[DN + c], 0.f);
    out[i] = v + x[i];
}

// ---------------------------------------------------------------------------
extern "C" void kernel_launch(void* const* d_in, const int* in_sizes, int n_in,
                              void* d_out, int out_size, void* d_ws, size_t ws_size,
                              hipStream_t stream) {
    (void)in_sizes; (void)n_in; (void)out_size; (void)ws_size;
    const float* x         = (const float*)d_in[0];
    const int*   eidx      = (const int*)  d_in[1];
    const float* edge_f    = (const float*)d_in[2];
    const float* edge_attr = (const float*)d_in[3];
    const float* lin1_w    = (const float*)d_in[4];
    const float* lin1_b    = (const float*)d_in[5];
    const float* lin2_w    = (const float*)d_in[6];
    const float* lin2_b    = (const float*)d_in[7];
    const float* linw_w    = (const float*)d_in[8];
    const float* linw_b    = (const float*)d_in[9];
    const float* gcn_w     = (const float*)d_in[10];
    const float* gcn_b     = (const float*)d_in[11];
    const float* bn_e_g    = (const float*)d_in[12];
    const float* bn_e_b    = (const float*)d_in[13];
    const float* bn_n_g    = (const float*)d_in[14];
    const float* bn_n_b    = (const float*)d_in[15];

    float* out_node = (float*)d_out;                                  // [N,128]
    float* efreg    = (float*)d_out + (size_t)N_NODES * DN;           // [E,64] scratch->final

    // workspace carve (256B aligned)
    char* p = (char*)d_ws;
    auto carve = [&](size_t bytes) -> void* {
        void* r = (void*)p;
        p += (bytes + 255) & ~(size_t)255;
        return r;
    };
    _Float16* x16 = (_Float16*)carve((size_t)N_NODES * DN * 2);
    _Float16* h16 = (_Float16*)carve((size_t)N_NODES * DN * 2);
    float*    xw  = (float*)   carve((size_t)N_NODES * DN * 4);
    float*    agg = (float*)   carve((size_t)N_NODES * DN * 4);
    float*    wE  = (float*)   carve((size_t)N_EDGES * 4);
    float*    deg = (float*)   carve((size_t)N_NODES * 4);
    float*    dnv = (float*)   carve((size_t)N_NODES * 4);
    _Float16* pb1 = (_Float16*)carve((size_t)4 * 8 * 32 * 16 * 2);    // lin1 packed
    _Float16* pbg = (_Float16*)carve((size_t)4 * 8 * 32 * 16 * 2);    // gcn packed
    _Float16* pb2 = (_Float16*)carve((size_t)7 * 4 * 32 * 16 * 2);    // lin2 packed
    float*    ebn = (float*)   carve(256 * 4);  // [sum64|sumsq64|scale64|shift64]
    float*    nbn = (float*)   carve(512 * 4);  // [sum128|sumsq128|scale128|shift128]

    const int NELT = N_NODES * DN;               // 6.4M
    // init accumulators (every call: graph replay safe)
    k_init<<<(NELT + 255) / 256, 256, 0, stream>>>(agg, deg, ebn, nbn);
    // f16 copies + packed weights
    k_cvt_f16<<<(NELT + 255) / 256, 256, 0, stream>>>(x, x16, NELT);
    k_pack_b<<<(4 * 8 * 32 + 63) / 64, 64, 0, stream>>>(lin1_w, pb1, DN, DN, 4, 8);
    k_pack_b<<<(4 * 8 * 32 + 63) / 64, 64, 0, stream>>>(gcn_w,  pbg, DN, DN, 4, 8);
    k_pack_b<<<(7 * 4 * 32 + 63) / 64, 64, 0, stream>>>(lin2_w, pb2, KIN, DE, 7, 4);
    // node GEMMs (h and xw)
    k_node_gemm<<<N_NODES / 16, 256, 0, stream>>>(x16, pb1, pbg, lin1_b, h16, xw);
    // edge GEMM -> ef_pre in d_out ef region
    k_edge_gemm<<<N_EDGES / 16, 128, 0, stream>>>(h16, eidx, edge_attr, edge_f, pb2,
                                                  lin2_b, efreg);
    // edge BN
    k_ebn_stats<<<512, 256, 0, stream>>>(efreg, ebn);
    k_ebn_final<<<1, 64, 0, stream>>>(ebn, bn_e_g, bn_e_b, ebn + 128);
    k_edge_pass2<<<(N_EDGES + 255) / 256, 256, 0, stream>>>(efreg, edge_attr, ebn + 128,
                                                            linw_w, linw_b, wE);
    // GCN norm + aggregation
    k_degree<<<(N_EDGES + 255) / 256, 256, 0, stream>>>(eidx, wE, deg);
    k_dinv<<<(N_NODES + 255) / 256, 256, 0, stream>>>(deg, dnv);
    k_scatter<<<(N_EDGES * 32 + 255) / 256, 256, 0, stream>>>(eidx, wE, dnv, xw, agg);
    // node BN + output
    k_node_pre<<<1024, 256, 0, stream>>>(agg, xw, dnv, gcn_b, nbn);
    k_nbn_final<<<1, 128, 0, stream>>>(nbn, bn_n_g, bn_n_b, nbn + 256);
    k_node_out<<<(NELT + 255) / 256, 256, 0, stream>>>(agg, x, nbn + 256, out_node);
}
